// GNN_67783173865504
// MI455X (gfx1250) — matrix operand order repaired and checked by reading, not compiled, not run on hardware
//
#include <hip/hip_runtime.h>
#include <hip/hip_bf16.h>
#include <math.h>

// Problem constants (from reference setup_inputs)
#define NN      50000      // nodes
#define NE      800000     // edges
#define DIN     128        // input/hidden feature dim
#define DOUT    64         // output feature dim
#define NGRAPH  64         // graphs in batch
#define NTILES  ((NN + 15) / 16)   // 3125 row tiles of 16 nodes

typedef _Float16 v16h __attribute__((ext_vector_type(16)));
typedef _Float16 v8h  __attribute__((ext_vector_type(8)));
typedef _Float16 v4h  __attribute__((ext_vector_type(4)));
typedef float    v8f  __attribute__((ext_vector_type(8)));
typedef float    v4f  __attribute__((ext_vector_type(4)));

// ---------------------------------------------------------------------------
// zero a float buffer
// ---------------------------------------------------------------------------
__global__ void zero_f32(float* __restrict__ p, int n) {
    int i = blockIdx.x * blockDim.x + threadIdx.x;
    if (i < n) p[i] = 0.0f;
}

// ---------------------------------------------------------------------------
// one-shot f32 -> f16 conversion (weights + input features)
// ---------------------------------------------------------------------------
__global__ void cvt_f32_to_f16(const float* __restrict__ in,
                               _Float16* __restrict__ outp, int n) {
    int i = blockIdx.x * blockDim.x + threadIdx.x;
    if (i < n) outp[i] = (_Float16)in[i];
}

// ---------------------------------------------------------------------------
// Edge aggregation: one wave (32 lanes) per edge, f16 feature gather.
// Lane l gathers feat[src][4l..4l+3] (8B), converts, scatter-adds f32 into
// agg[dst]. Optionally accumulates in-degree (layer 1 only). Prefetches the
// source row of a far-ahead edge (global_prefetch_b8 path).
// ---------------------------------------------------------------------------
__global__ __launch_bounds__(256) void edge_agg_f16(
    const _Float16* __restrict__ feat,
    const int*      __restrict__ src,
    const int*      __restrict__ dst,
    float*          __restrict__ agg,
    float*          __restrict__ cnt)   // nullptr -> skip degree counting
{
    int wave = threadIdx.x >> 5;
    int lane = threadIdx.x & 31;
    int e = blockIdx.x * (blockDim.x >> 5) + wave;
    if (e >= NE) return;

    int s = src[e];
    int d = dst[e];

    // software prefetch of a far-ahead source row (2 cachelines of f16 row)
    int e2 = e + 2048;
    if (e2 < NE && lane < 2) {
        int s2 = src[e2];
        __builtin_prefetch(feat + (size_t)s2 * DIN + lane * 64, 0, 1);
    }

    v4h v = *(const v4h*)(feat + (size_t)s * DIN + lane * 4);
    float* arow = agg + (size_t)d * DIN + lane * 4;
    atomicAdd(arow + 0, (float)v[0]);
    atomicAdd(arow + 1, (float)v[1]);
    atomicAdd(arow + 2, (float)v[2]);
    atomicAdd(arow + 3, (float)v[3]);

    if (cnt != nullptr && lane == 0) atomicAdd(cnt + d, 1.0f);
}

// ---------------------------------------------------------------------------
// Fragment loaders (CDNA5 wave32 WMMA layouts, 16-bit operands)
//
// A (16x32, MxK): lane l holds row m = l&15.
//   lanes 0-15 : elems 0-15 -> K = kc*32 + 0..15 (contiguous, same order as B)
//   lanes 16-31: elems 0-15 -> K = kc*32 + 16..31
//
// B (32x16, KxN): lane l holds column n = l&15.
//   lanes 0-15 hold K = kc*32 + 0..15, lanes 16-31 hold K = kc*32 + 16..31
// ---------------------------------------------------------------------------

// A fragment from f32 source with per-row scale (mean path): scale + convert
__device__ inline v16h load_frag_A_f32(const float* __restrict__ rowk, float scale) {
    v4f a0 = *(const v4f*)(rowk + 0);
    v4f a1 = *(const v4f*)(rowk + 4);
    v4f a2 = *(const v4f*)(rowk + 8);
    v4f a3 = *(const v4f*)(rowk + 12);
    v16h r;
#pragma unroll
    for (int j = 0; j < 4; ++j) {
        r[j]      = (_Float16)(a0[j] * scale);
        r[4 + j]  = (_Float16)(a1[j] * scale);
        r[8 + j]  = (_Float16)(a2[j] * scale);
        r[12 + j] = (_Float16)(a3[j] * scale);
    }
    return r;
}

// A fragment straight from f16 source (root path): two 16B loads + shuffle
__device__ inline v16h load_frag_A_f16(const _Float16* __restrict__ rowk) {
    v8h lo = *(const v8h*)(rowk + 0);    // K = kb .. kb+7
    v8h hi = *(const v8h*)(rowk + 8);    // K = kb+8 .. kb+15
    return __builtin_shufflevector(lo, hi, 0, 1, 2, 3, 4, 5, 6, 7,
                                           8, 9, 10, 11, 12, 13, 14, 15);
}

// B fragment from LDS-staged f16 weights: one 32B LDS read (2x ds_load_b128)
__device__ inline v16h load_frag_B_lds(const _Float16* rowk) {
    return *(const v16h*)rowk;
}

__device__ inline float gelu_exact(float x) {
    return 0.5f * x * (1.0f + erff(x * 0.70710678118654752f));
}

// ---------------------------------------------------------------------------
// Fused SAGEConv node kernel:
//   out[m][n] = GELU( mean[m] @ Wl^T + bias + act[m] @ Wr^T )
// One wave per 16-node row tile; OUT_TILES column tiles of 16.
// K = 128 consumed as 4 chunks of 32 via v_wmma_f32_16x16x32_f16.
// Weights are staged once per workgroup into dynamic LDS; the hot loop feeds
// WMMA from ds_load_b128 (8 waves share one LDS copy instead of 8x global).
// POOL=false: write f16 node features (layer 1 -> hh).
// POOL=true : atomic global_add_pool into pool_out[batch[m]][n] (layer 2).
// ---------------------------------------------------------------------------
template <int OUT_TILES, bool POOL>
__global__ __launch_bounds__(256) void node_layer(
    const float*    __restrict__ agg,    // [N][128] f32 neighbor sums
    const float*    __restrict__ cnt,    // [N] degrees
    const _Float16* __restrict__ actx,   // [N][128] f16 root features
    const _Float16* __restrict__ Wl,     // [OUT_TILES*16][128] f16
    const float*    __restrict__ bias,   // [OUT_TILES*16] f32
    const _Float16* __restrict__ Wr,     // [OUT_TILES*16][128] f16
    _Float16*       __restrict__ hout,      // POOL=false: [N][OUT_TILES*16] f16
    float*          __restrict__ pool_out,  // POOL=true : [NGRAPH][OUT_TILES*16]
    const int*      __restrict__ batch)
{
    extern __shared__ _Float16 smem[];           // [2][OUT_TILES*16][128]
    _Float16* wl_lds = smem;
    _Float16* wr_lds = smem + OUT_TILES * 16 * DIN;

    // --- stage both weight matrices into LDS (cooperative, 16B chunks) ---
    const int vec_total = OUT_TILES * 16 * DIN / 8;   // v8h chunks per matrix
    for (int i = threadIdx.x; i < vec_total; i += 256) {
        *(v8h*)(wl_lds + i * 8) = *(const v8h*)(Wl + i * 8);
        *(v8h*)(wr_lds + i * 8) = *(const v8h*)(Wr + i * 8);
    }
    __syncthreads();

    const int wave = threadIdx.x >> 5;
    const int lane = threadIdx.x & 31;
    const int tile = blockIdx.x * 8 + wave;

    if (tile < NTILES) {                      // wave-uniform: EXEC stays all-1s
        const int m0  = tile * 16;
        const int lm  = lane & 15;
        const int kbA = (lane < 16) ? 0 : 16; // A-fragment K sub-base
        const int kbB = (lane < 16) ? 0 : 16; // B-fragment K sub-base

        const int   node_a = m0 + lm;         // row this lane supplies to A
        const float deg    = cnt[node_a];
        const float invdeg = 1.0f / fmaxf(deg, 1.0f);
        const float*    aggrow = agg  + (size_t)node_a * DIN;
        const _Float16* xrow   = actx + (size_t)node_a * DIN;

        // C/D layout: lane holds col n = lm; acc elem r holds row m0 + mb + r
        v8f acc[OUT_TILES];
#pragma unroll
        for (int nt = 0; nt < OUT_TILES; ++nt) {
            float b = bias[nt * 16 + lm];
#pragma unroll
            for (int r = 0; r < 8; ++r) acc[nt][r] = b;
        }

#pragma unroll
        for (int kc = 0; kc < DIN / 32; ++kc) {
            v16h Am = load_frag_A_f32(aggrow + kc * 32 + kbA, invdeg); // mean
            v16h Ax = load_frag_A_f16(xrow   + kc * 32 + kbA);         // root
#pragma unroll
            for (int nt = 0; nt < OUT_TILES; ++nt) {
                const int woff = (nt * 16 + lm) * DIN + kc * 32 + kbB;
                v16h Bl = load_frag_B_lds(wl_lds + woff);
                v16h Br = load_frag_B_lds(wr_lds + woff);
                acc[nt] = __builtin_amdgcn_wmma_f32_16x16x32_f16(
                    false, Am, false, Bl, (short)0, acc[nt], false, false);
                acc[nt] = __builtin_amdgcn_wmma_f32_16x16x32_f16(
                    false, Ax, false, Br, (short)0, acc[nt], false, false);
            }
        }

        // Epilogue: GELU + write f16 features / atomic pool
        const int mb = (lane < 16) ? 0 : 8;
#pragma unroll
        for (int nt = 0; nt < OUT_TILES; ++nt) {
            const int n = nt * 16 + lm;
#pragma unroll
            for (int r = 0; r < 8; ++r) {
                float v = gelu_exact(acc[nt][r]);
                int node = m0 + mb + r;
                if (node < NN) {
                    if (POOL) {
                        int g = batch[node];
                        atomicAdd(pool_out + (size_t)g * (OUT_TILES * 16) + n, v);
                    } else {
                        hout[(size_t)node * (OUT_TILES * 16) + n] = (_Float16)v;
                    }
                }
            }
        }
    }
}

// ---------------------------------------------------------------------------
// Orchestration.
// Inputs (setup_inputs order):
//  0: x        [N,128] f32      1: edge_index [2,E] i32   2: batch [N] i32
//  3: W1l [128,128]  4: b1 [128]  5: W1r [128,128]
//  6: W2l [64,128]   7: b2 [64]   8: W2r [64,128]
// Output: [64,64] f32 (global_add_pool of layer-2 GELU features)
// Workspace: cnt [N] f32 | agg [N,128] f32 | xh [N,128] f16 | hh [N,128] f16 |
//            W1l/W1r/W2l/W2r f16 (98KB)                          (~51.5 MB)
// ---------------------------------------------------------------------------
extern "C" void kernel_launch(void* const* d_in, const int* in_sizes, int n_in,
                              void* d_out, int out_size, void* d_ws, size_t ws_size,
                              hipStream_t stream) {
    const float* x   = (const float*)d_in[0];
    const int*   ei  = (const int*)d_in[1];
    const int*   bat = (const int*)d_in[2];
    const float* W1l = (const float*)d_in[3];
    const float* b1  = (const float*)d_in[4];
    const float* W1r = (const float*)d_in[5];
    const float* W2l = (const float*)d_in[6];
    const float* b2  = (const float*)d_in[7];
    const float* W2r = (const float*)d_in[8];
    float* out = (float*)d_out;

    const int* src = ei;        // edge_index[0]
    const int* dst = ei + NE;   // edge_index[1]

    // Workspace carve-up (256B aligned blocks)
    char* ws = (char*)d_ws;
    size_t off = 0;
    auto carve = [&](size_t bytes) {
        void* p = ws + off;
        off += (bytes + 255) & ~(size_t)255;
        return p;
    };
    float*     cnt  = (float*)    carve((size_t)NN * 4);
    float*     agg  = (float*)    carve((size_t)NN * DIN * 4);
    _Float16*  xh   = (_Float16*) carve((size_t)NN * DIN * 2);
    _Float16*  hh   = (_Float16*) carve((size_t)NN * DIN * 2);
    _Float16*  w1lh = (_Float16*) carve((size_t)DIN * DIN * 2);
    _Float16*  w1rh = (_Float16*) carve((size_t)DIN * DIN * 2);
    _Float16*  w2lh = (_Float16*) carve((size_t)DOUT * DIN * 2);
    _Float16*  w2rh = (_Float16*) carve((size_t)DOUT * DIN * 2);

    const int ZT = 256;
    const int waves_per_block = 8;
    const int edge_blocks = (NE + waves_per_block - 1) / waves_per_block;
    const int node_blocks = (NTILES + waves_per_block - 1) / waves_per_block;
    const size_t lds1 = (size_t)2 * DIN  * DIN * 2;  // 64KB: W1l+W1r f16
    const size_t lds2 = (size_t)2 * DOUT * DIN * 2;  // 32KB: W2l+W2r f16

    // 0) pre-convert input features and weights to f16
    cvt_f32_to_f16<<<((NN * DIN) + ZT - 1) / ZT, ZT, 0, stream>>>(x, xh, NN * DIN);
    cvt_f32_to_f16<<<((DIN * DIN) + ZT - 1) / ZT, ZT, 0, stream>>>(W1l, w1lh, DIN * DIN);
    cvt_f32_to_f16<<<((DIN * DIN) + ZT - 1) / ZT, ZT, 0, stream>>>(W1r, w1rh, DIN * DIN);
    cvt_f32_to_f16<<<((DOUT * DIN) + ZT - 1) / ZT, ZT, 0, stream>>>(W2l, w2lh, DOUT * DIN);
    cvt_f32_to_f16<<<((DOUT * DIN) + ZT - 1) / ZT, ZT, 0, stream>>>(W2r, w2rh, DOUT * DIN);

    // 1) zero degree counts, aggregation buffer, and pooled output
    zero_f32<<<(NN + ZT - 1) / ZT, ZT, 0, stream>>>(cnt, NN);
    zero_f32<<<((NN * DIN) + ZT - 1) / ZT, ZT, 0, stream>>>(agg, NN * DIN);
    zero_f32<<<(NGRAPH * DOUT + ZT - 1) / ZT, ZT, 0, stream>>>(out, NGRAPH * DOUT);

    // 2) layer-1 scatter aggregation (+ degree)
    edge_agg_f16<<<edge_blocks, 256, 0, stream>>>(xh, src, dst, agg, cnt);

    // 3) layer-1 fused SAGEConv + GELU -> hh (f16)
    node_layer<8, false><<<node_blocks, 256, lds1, stream>>>(
        agg, cnt, xh, w1lh, b1, w1rh, hh, nullptr, nullptr);

    // 4) re-zero aggregation buffer
    zero_f32<<<((NN * DIN) + ZT - 1) / ZT, ZT, 0, stream>>>(agg, NN * DIN);

    // 5) layer-2 scatter aggregation (degrees unchanged)
    edge_agg_f16<<<edge_blocks, 256, 0, stream>>>(hh, src, dst, agg, nullptr);

    // 6) layer-2 fused SAGEConv + GELU + global_add_pool -> out
    node_layer<4, true><<<node_blocks, 256, lds2, stream>>>(
        agg, cnt, hh, w2lh, b2, w2rh, nullptr, out, bat);
}
